// LocalAttention_45526653337810
// MI455X (gfx1250) — matrix operand order repaired and checked by reading, not compiled, not run on hardware
//
#include <hip/hip_runtime.h>

// ---- CDNA5 WMMA / vector types ---------------------------------------------
typedef __attribute__((ext_vector_type(16))) __bf16         v16bf;
typedef __attribute__((ext_vector_type(8)))  unsigned short v8us;
typedef __attribute__((ext_vector_type(8)))  float          v8f;
typedef __attribute__((ext_vector_type(4)))  unsigned int   u32x4;
typedef __attribute__((ext_vector_type(8)))  int            i32x8;
typedef __attribute__((ext_vector_type(4)))  int            i32x4;

struct bfrag { v8us lo, hi; };   // 32 bytes -> bit_cast to v16bf

// Native f32->bf16 (RNE): lowers to v_cvt_pk_bf16_f32 on gfx1250.
__device__ __forceinline__ unsigned short f2bf(float f) {
    __bf16 h = (__bf16)f;
    return __builtin_bit_cast(unsigned short, h);
}

__device__ __forceinline__ v8f wmma_bf16(v16bf a, v16bf b, v8f c) {
    return __builtin_amdgcn_wmma_f32_16x16x32_bf16(
        false, a, false, b, (short)0, c, false, false);
}

// A-fragment (16x32 bf16) from row-major [m][k] storage.
// Lane: m = m0 + lane%16, kh = (lane>=16)?8:0.
// elems 0..7  = k0+kh+{0..7}   (16B contiguous -> ds_load_b128)
// elems 8..15 = k0+kh+16+{0..7}(16B contiguous -> ds_load_b128)
__device__ __forceinline__ v16bf frag_a_rm(const unsigned short* p, int m0,
                                           int k0, int stride, int lane) {
    const int m  = m0 + (lane & 15);
    const int kh = (lane >= 16) ? 8 : 0;
    const unsigned short* q = p + m * stride + k0 + kh;
    bfrag f;
    f.lo = *(const v8us*)(q);
    f.hi = *(const v8us*)(q + 16);
    return __builtin_bit_cast(v16bf, f);
}

// B-fragment (32x16 bf16) from [n][k] storage.
// Lane: n = n0 + lane%16, kh = (lane>=16)?8:0; elems 0..7 = k0+kh+{0..7}, 8..15 = k0+kh+16+{0..7}.
__device__ __forceinline__ v16bf frag_b_nk(const unsigned short* p, int n0,
                                           int k0, int stride, int lane) {
    const int n  = n0 + (lane & 15);
    const int kh = (lane >= 16) ? 8 : 0;
    const unsigned short* q = p + n * stride + k0 + kh;
    bfrag f;
    f.lo = *(const v8us*)(q);
    f.hi = *(const v8us*)(q + 16);
    return __builtin_bit_cast(v16bf, f);
}

// ---- LDS layout (57600 B static; single block assumed at LDS offset 0) ------
struct Smem {
    unsigned short Xb[64][64];   // x window bf16, row-major [m][c]       8 KB
    unsigned short Wqb[16][64];  // Wq [o][c] = [n][k], rows 8..15 zero   2 KB
    unsigned short Wkb[16][64];  //                                       2 KB
    unsigned short Wvb[64][64];  // Wv [o][c] = [n][k]                    8 KB
    unsigned short qb[64][32];   // q  [m][cq], cols 8..31 zero           4 KB
    unsigned short kb[64][32];   // k  [m][cq] == B-operand [n][k]        4 KB
    unsigned short Pb[64][64];   // softmax  [m][m'], pad zero            8 KB
    unsigned short Vc[64][64];   // v channel-major [c][m'] == [n][k]     8 KB
    float          Xraw[64 * 49];// TDM tile: [c][m] f32 (residual)    12.5 KB
};
// Epilogue stage Of2 = float[64][56] (14336 B) aliases Xb/Wqb/Wkb/head(Wvb),
// all dead after phase 1 (barriers separate last uses). Xraw/Pb/Vc untouched.

__global__ __launch_bounds__(128)
void local_attn_fused(const float* __restrict__ x,
                      const float* __restrict__ Wq, const float* __restrict__ bq,
                      const float* __restrict__ Wk, const float* __restrict__ bk,
                      const float* __restrict__ Wv, const float* __restrict__ bv,
                      const float* __restrict__ gamma,
                      float* __restrict__ out) {
    __shared__ __align__(32) unsigned char smem_raw[sizeof(Smem)];
    Smem* S = reinterpret_cast<Smem*>(smem_raw);
    float (*Of2)[56] = reinterpret_cast<float(*)[56]>(smem_raw);  // epilogue alias

    const int t    = threadIdx.x;
    const int lane = t & 31;
    const int wave = t >> 5;
    const int hi8  = (lane >= 16) ? 8 : 0;
    const int mrow = wave * 16;

    const int blk = blockIdx.x;
    const int b   = blk >> 10;
    const int l   = blk & 1023;
    const int h0  = (l >> 5) * 7;
    const int w0  = (l & 31) * 7;

    const float gm = gamma[0];

    // ---------------- Phase 0: async tensor stage of x window ----------------
#if __has_builtin(__builtin_amdgcn_tensor_load_to_lds)
    if (t < 32) {
        // D# per CDNA5 ISA ch.8: 3D tile x=7 (contig), y=7 (stride W), z=64 (stride H*W)
        const unsigned long long ga = (unsigned long long)(uintptr_t)x +
            4ull * ((((unsigned long long)b * 64) * 224 + h0) * 224 + w0);
        const unsigned ldsOff = (unsigned)offsetof(Smem, Xraw);
        u32x4 g0;
        g0[0] = 1u;                                         // count=1, user desc
        g0[1] = ldsOff;                                     // lds_addr (bytes)
        g0[2] = (unsigned)(ga & 0xFFFFFFFFull);             // global_addr[31:0]
        g0[3] = (unsigned)((ga >> 32) & 0x1FFFFFFull)       // global_addr[56:32]
              | (2u << 30);                                 // type = 2 ("image")
        i32x8 g1;
        g1[0] = (int)(2u << 16);                            // data_size=4B, mask=0
        g1[1] = (int)((224u & 0xFFFFu) << 16);              // tensor_dim0 lo16 @bit48
        g1[2] = (int)((224u >> 16) | ((224u & 0xFFFFu) << 16)); // dim0 hi | dim1 lo
        g1[3] = (int)((224u >> 16) | (7u << 16));           // dim1 hi | tile_dim0=7
        g1[4] = (int)(7u | (64u << 16));                    // tile_dim1=7, tile_dim2=64
        g1[5] = (int)224u;                                  // tensor_dim0_stride lo32
        g1[6] = (int)((50176u & 0xFFFFu) << 16);            // str0 hi16 | str1 lo16
        g1[7] = (int)(50176u >> 16);                        // tensor_dim1_stride[47:16]
        i32x4 g2 = { 64, 0, 0, 0 };                         // tensor_dim2=64
        i32x4 g3 = { 0, 0, 0, 0 };
        i32x8 g4 = { 0, 0, 0, 0, 0, 0, 0, 0 };              // VADDR4 group: unused
        __builtin_amdgcn_tensor_load_to_lds(g0, g1, g2, g3, g4, 0);
    }
#else
    for (int i = t; i < 49 * 64; i += 128) {
        const int n = i % 49, c = i / 49;
        S->Xraw[c * 49 + n] =
            x[(((size_t)b * 64 + c) * 224 + (h0 + n / 7)) * 224 + (w0 + n % 7)];
    }
#endif

    // Weight staging (direct [n][k] copies, coalesced) overlaps the TDM.
    for (int i = t; i < 16 * 64; i += 128) {
        const int o = i >> 6, c = i & 63;
        S->Wqb[o][c] = (o < 8) ? f2bf(Wq[o * 64 + c]) : (unsigned short)0;
        S->Wkb[o][c] = (o < 8) ? f2bf(Wk[o * 64 + c]) : (unsigned short)0;
    }
    for (int i = t; i < 64 * 64; i += 128) {
        const int o = i >> 6, c = i & 63;
        S->Wvb[o][c] = f2bf(Wv[o * 64 + c]);
    }
    for (int i = t; i < 64 * 16; i += 128) {       // zero q/k pad columns 16..31
        const int r = i >> 4, c = 16 + (i & 15);
        S->qb[r][c] = 0;  S->kb[r][c] = 0;
    }
    for (int i = t; i < 15 * 64; i += 128) {       // zero pad rows 49..63
        const int r = 49 + (i >> 6), c = i & 63;
        S->Xb[r][c] = 0;  S->Pb[r][c] = 0;
    }
#if __has_builtin(__builtin_amdgcn_tensor_load_to_lds)
    if (t < 32) __builtin_amdgcn_s_wait_tensorcnt(0);
#endif
    __syncthreads();

    // LDS transform: Xraw [c][m] f32 -> Xb [m][c] bf16 (WMMA A layout)
    for (int i = t; i < 49 * 64; i += 128) {
        const int n = i % 49, c = i / 49;
        S->Xb[n][c] = f2bf(S->Xraw[c * 49 + n]);
    }
    __syncthreads();

    // ---------------- Phase 1: Q, K, V projections ----------------
    const v16bf xa0 = frag_a_rm(&S->Xb[0][0], mrow, 0,  64, lane);
    const v16bf xa1 = frag_a_rm(&S->Xb[0][0], mrow, 32, 64, lane);
    {
        v8f acc = {};
        acc = wmma_bf16(xa0, frag_b_nk(&S->Wqb[0][0], 0, 0,  64, lane), acc);
        acc = wmma_bf16(xa1, frag_b_nk(&S->Wqb[0][0], 0, 32, 64, lane), acc);
        const int n = lane & 15;
        const float bias = (n < 8) ? bq[n] : 0.0f;
#pragma unroll
        for (int v = 0; v < 8; ++v) S->qb[mrow + hi8 + v][n] = f2bf(acc[v] + bias);
    }
    {
        v8f acc = {};
        acc = wmma_bf16(xa0, frag_b_nk(&S->Wkb[0][0], 0, 0,  64, lane), acc);
        acc = wmma_bf16(xa1, frag_b_nk(&S->Wkb[0][0], 0, 32, 64, lane), acc);
        const int n = lane & 15;
        const float bias = (n < 8) ? bk[n] : 0.0f;
#pragma unroll
        for (int v = 0; v < 8; ++v) S->kb[mrow + hi8 + v][n] = f2bf(acc[v] + bias);
    }
#pragma unroll
    for (int nt = 0; nt < 4; ++nt) {
        v8f acc = {};
        acc = wmma_bf16(xa0, frag_b_nk(&S->Wvb[0][0], nt * 16, 0,  64, lane), acc);
        acc = wmma_bf16(xa1, frag_b_nk(&S->Wvb[0][0], nt * 16, 32, 64, lane), acc);
        const int c = nt * 16 + (lane & 15);
        const float bias = bv[c];
        v8us pack;
#pragma unroll
        for (int v = 0; v < 8; ++v) pack[v] = f2bf(acc[v] + bias);
        *(v8us*)&S->Vc[c][mrow + hi8] = pack;      // contiguous 16B store
    }
    __syncthreads();

    // ---------------- Phase 2: S = q k^T ; softmax in registers ----------------
    const v16bf qa = frag_a_rm(&S->qb[0][0], mrow, 0, 32, lane);
    v8f s[4];
#pragma unroll
    for (int nt = 0; nt < 4; ++nt) {
        v8f acc = {};
        acc = wmma_bf16(qa, frag_b_nk(&S->kb[0][0], nt * 16, 0, 32, lane), acc);
        s[nt] = acc;
    }
#pragma unroll
    for (int v = 0; v < 8; ++v) {
        const int m = mrow + hi8 + v;
        float vals[4];
        float mx = -__builtin_inff();
#pragma unroll
        for (int nt = 0; nt < 4; ++nt) {
            const int n = nt * 16 + (lane & 15);
            vals[nt] = (n < 49) ? s[nt][v] : -__builtin_inff();
            mx = fmaxf(mx, vals[nt]);
        }
#pragma unroll
        for (int off = 1; off < 16; off <<= 1)
            mx = fmaxf(mx, __shfl_xor(mx, off, 32));
        float sum = 0.0f;
#pragma unroll
        for (int nt = 0; nt < 4; ++nt) {
            const int n = nt * 16 + (lane & 15);
            const float e = (n < 49) ? __expf(vals[nt] - mx) : 0.0f;
            vals[nt] = e;
            sum += e;
        }
#pragma unroll
        for (int off = 1; off < 16; off <<= 1)
            sum += __shfl_xor(sum, off, 32);
        const float inv = 1.0f / sum;
        if (m < 49) {
#pragma unroll
            for (int nt = 0; nt < 4; ++nt)
                S->Pb[m][nt * 16 + (lane & 15)] = f2bf(vals[nt] * inv);
        }
    }
    __syncthreads();

    // ---------------- Phase 3: O = P V ; epilogue gamma*O + x ----------------
    const v16bf pa0 = frag_a_rm(&S->Pb[0][0], mrow, 0,  64, lane);
    const v16bf pa1 = frag_a_rm(&S->Pb[0][0], mrow, 32, 64, lane);
#pragma unroll
    for (int nt = 0; nt < 4; ++nt) {
        v8f acc = {};
        acc = wmma_bf16(pa0, frag_b_nk(&S->Vc[0][0], nt * 16, 0,  64, lane), acc);
        acc = wmma_bf16(pa1, frag_b_nk(&S->Vc[0][0], nt * 16, 32, 64, lane), acc);
        const int c = nt * 16 + (lane & 15);
        v8f res;
#pragma unroll
        for (int v = 0; v < 8; ++v) {
            const int m = mrow + hi8 + v;
            res[v] = gm * acc[v] + ((m < 49) ? S->Xraw[c * 49 + m] : 0.0f);
        }
        *(v8f*)&Of2[c][mrow + hi8] = res;          // contiguous 32B store
    }
    __syncthreads();

    // coalesced writeout (contiguous runs of 7 floats in w)
    for (int i = t; i < 49 * 64; i += 128) {
        const int n = i % 49, c = i / 49;
        out[(((size_t)b * 64 + c) * 224 + (h0 + n / 7)) * 224 + (w0 + n % 7)] =
            Of2[c][n];
    }
}

extern "C" void kernel_launch(void* const* d_in, const int* in_sizes, int n_in,
                              void* d_out, int out_size, void* d_ws, size_t ws_size,
                              hipStream_t stream) {
    const float* x     = (const float*)d_in[0];
    const float* Wq    = (const float*)d_in[1];
    const float* bq    = (const float*)d_in[2];
    const float* Wk    = (const float*)d_in[3];
    const float* bk    = (const float*)d_in[4];
    const float* Wv    = (const float*)d_in[5];
    const float* bv    = (const float*)d_in[6];
    const float* gamma = (const float*)d_in[7];
    float* out = (float*)d_out;

    const int nWindows = 16 * 32 * 32;   // B * (H/7) * (W/7)
    local_attn_fused<<<dim3(nWindows), dim3(128), 0, stream>>>(
        x, Wq, bq, Wk, bk, Wv, bv, gamma, out);
}
